// GRUSimple_78658031059231
// MI455X (gfx1250) — matrix operand order repaired and checked by reading, not compiled, not run on hardware
//
#include <hip/hip_runtime.h>
#include <hip/hip_bf16.h>
#include <stdint.h>

#define B_  128
#define S_  1024
#define F_  64
#define H_  512
#define O_  64
#define K1  192     // 3*F
#define N3  1536    // 3*H

typedef __attribute__((ext_vector_type(16))) _Float16 v16h;
typedef __attribute__((ext_vector_type(8)))  float    v8f;

union Frag16 { v16h h; unsigned u[8]; };

// ---------------------------------------------------------------- helpers
static __device__ __forceinline__ unsigned ldsOff(const void* p) {
  // flat LDS aperture: low 32 bits == LDS byte address (ISA 10.2)
  return (unsigned)(reinterpret_cast<uintptr_t>(p));
}

static __device__ __forceinline__ void asyncLoad16(unsigned lds, const void* g) {
  asm volatile("global_load_async_to_lds_b128 %0, %1, off"
               :: "v"(lds), "v"((unsigned long long)(uintptr_t)g) : "memory");
}

static __device__ __forceinline__ void waitAsync() {
  asm volatile("s_wait_asynccnt 0" ::: "memory");
}

static __device__ __forceinline__ v8f splat8(float x) {
  v8f r;
#pragma unroll
  for (int e = 0; e < 8; ++e) r[e] = x;
  return r;
}

static __device__ __forceinline__ float sigm(float x) {
  return 1.0f / (1.0f + __expf(-x));
}

// Fused 3-gate GEMM: one shared A fragment per K-tile feeds three WMMAs
// (gate column blocks n0, n0+512, n0+1024 of the [N][K] row-major weight).
// A comes from LDS (row m = lo, stride lda halfs); accumulators in f32.
static __device__ __forceinline__ void mm3(const _Float16* __restrict__ aLds, int lda,
                                           const _Float16* __restrict__ wG,  int ldw,
                                           int kTiles, int n0, int lo, int hi,
                                           v8f& accR, v8f& accZ, v8f& accN)
{
  const _Float16* aRow = aLds + lo * lda;
  const _Float16* bR = wG + (size_t)(n0)        * ldw;
  const _Float16* bZ = wG + (size_t)(n0 + 512)  * ldw;
  const _Float16* bN = wG + (size_t)(n0 + 1024) * ldw;
  for (int kk = 0; kk < kTiles; ++kk) {
    const int kb = kk * 32;
    Frag16 a, b0, b1, b2;
#pragma unroll
    for (int v = 0; v < 8; ++v) {
      const int ka = kb + 2 * v + 16 * hi;                        // A layout (contiguous k, matches B)
      a.u[v]  = *reinterpret_cast<const unsigned*>(aRow + ka);
      const int kv = kb + 2 * v + 16 * hi;                        // B layout
      b0.u[v] = *reinterpret_cast<const unsigned*>(bR + kv);
      b1.u[v] = *reinterpret_cast<const unsigned*>(bZ + kv);
      b2.u[v] = *reinterpret_cast<const unsigned*>(bN + kv);
    }
    accR = __builtin_amdgcn_wmma_f32_16x16x32_f16(false, a.h, false, b0.h,
                                                  (short)0, accR, false, false);
    accZ = __builtin_amdgcn_wmma_f32_16x16x32_f16(false, a.h, false, b1.h,
                                                  (short)0, accZ, false, false);
    accN = __builtin_amdgcn_wmma_f32_16x16x32_f16(false, a.h, false, b2.h,
                                                  (short)0, accN, false, false);
  }
}

// ---------------------------------------------------------------- kernel 1
// forward-fill imputation + pack [x_imp | mask | delta] as f16 [B,S,192]
__global__ __launch_bounds__(256)
void impute_pack(const float* __restrict__ x, const float* __restrict__ mask,
                 const float* __restrict__ delta, const float* __restrict__ xmean,
                 _Float16* __restrict__ cmb)
{
  const int id = blockIdx.x * blockDim.x + threadIdx.x;   // 0 .. B*F-1
  if (id >= B_ * F_) return;
  const int b = id / F_, f = id % F_;
  float carry = xmean[f];
  for (int t = 0; t < S_; ++t) {
    const size_t src = ((size_t)b * S_ + t) * F_ + f;
    const float xt = x[src], mt = mask[src], dt = delta[src];
    carry = (mt > 0.0f) ? xt : carry;
    const float imp = (mt == 0.0f) ? carry : xt;
    const size_t dst = ((size_t)b * S_ + t) * K1;
    cmb[dst + f]          = (_Float16)imp;
    cmb[dst + F_ + f]     = (_Float16)mt;
    cmb[dst + 2 * F_ + f] = (_Float16)dt;
  }
}

// ---------------------------------------------------------------- kernel 2
__global__ __launch_bounds__(256)
void cvt_weights(const float* __restrict__ Wih, const float* __restrict__ Whh,
                 _Float16* __restrict__ wih16, _Float16* __restrict__ whh16)
{
  const int NIH = N3 * K1;            // 294912
  const int NHH = N3 * H_;            // 786432
  const int id = blockIdx.x * blockDim.x + threadIdx.x;
  if (id < NIH)            wih16[id]       = (_Float16)Wih[id];
  else if (id < NIH + NHH) whh16[id - NIH] = (_Float16)Whh[id - NIH];
}

// ---------------------------------------------------------------- kernel 3
// One workgroup per 16-row batch tile; h lives in LDS; no cross-WG sync.
static __device__ __forceinline__ void issueCmbTile(const _Float16* __restrict__ cmb,
                                                    _Float16* dstLds, int b0, int t, int tid)
{
  // 16 rows x 384B = 384 chunks of 16B
  {
    const int c = tid, row = c / 24, off = (c % 24) * 8;
    asyncLoad16(ldsOff(dstLds + row * K1 + off),
                cmb + ((size_t)(b0 + row) * S_ + t) * K1 + off);
  }
  if (tid < 128) {
    const int c = tid + 256, row = c / 24, off = (c % 24) * 8;
    asyncLoad16(ldsOff(dstLds + row * K1 + off),
                cmb + ((size_t)(b0 + row) * S_ + t) * K1 + off);
  }
}

__global__ __launch_bounds__(256)
void grud_recurrence(const _Float16* __restrict__ cmb,
                     const _Float16* __restrict__ wih,
                     const _Float16* __restrict__ whh,
                     const float* __restrict__ bih,
                     const float* __restrict__ bhh,
                     float* __restrict__ hfin)
{
  __shared__ __align__(16) _Float16 hA[16 * H_];       // f16 copy of h (WMMA A operand)
  __shared__ __align__(16) float    hF[16 * H_];       // f32 master h
  __shared__ __align__(16) _Float16 cT[2][16 * K1];    // double-buffered combined_t tile

  const int tid  = threadIdx.x;
  const int lane = tid & 31;
  const int lo   = lane & 15;
  const int hi   = lane >> 4;
  const int w    = tid >> 5;            // wave 0..7
  const int b0   = blockIdx.x * 16;

  for (int i = tid; i < 16 * H_; i += 256) { hF[i] = 0.0f; hA[i] = (_Float16)0.0f; }

  issueCmbTile(cmb, cT[0], b0, 0, tid);
  waitAsync();
  __syncthreads();

  for (int t = 0; t < S_; ++t) {
    if (t + 1 < S_) issueCmbTile(cmb, cT[(t + 1) & 1], b0, t + 1, tid);
    const _Float16* cTt = cT[t & 1];

    float hn[4][8];
#pragma unroll
    for (int jj = 0; jj < 4; ++jj) {
      const int jc  = w + jj * 8;       // hidden column tile 0..31
      const int col = jc * 16 + lo;
      const int nR  = jc * 16 + lo;     // r-gate W row for this lane
      const int nZ  = nR + 512;
      const int nN  = nR + 1024;

      v8f iR = splat8(bih[nR]);
      v8f iZ = splat8(bih[nZ]);
      v8f iN = splat8(bih[nN]);
      mm3(cTt, K1, wih, K1, 6, nR, lo, hi, iR, iZ, iN);

      v8f hR = splat8(bhh[nR]);
      v8f hZ = splat8(bhh[nZ]);
      v8f hN = splat8(bhh[nN]);
      mm3(hA, H_, whh, H_, 16, nR, lo, hi, hR, hZ, hN);

#pragma unroll
      for (int e = 0; e < 8; ++e) {
        const float r  = sigm(iR[e] + hR[e]);
        const float z  = sigm(iZ[e] + hZ[e]);
        const float nn = tanhf(iN[e] + r * hN[e]);
        const float hp = hF[(e + 8 * hi) * H_ + col];
        hn[jj][e] = (1.0f - z) * nn + z * hp;
      }
    }

    __syncthreads();                    // everyone done reading hA/hF for step t
#pragma unroll
    for (int jj = 0; jj < 4; ++jj) {
      const int col = (w + jj * 8) * 16 + lo;
#pragma unroll
      for (int e = 0; e < 8; ++e) {
        const int m = e + 8 * hi;
        hF[m * H_ + col] = hn[jj][e];
        hA[m * H_ + col] = (_Float16)hn[jj][e];
      }
    }
    waitAsync();                        // next combined tile landed in LDS
    __syncthreads();
  }

  for (int i = tid; i < 16 * H_; i += 256) {
    const int r = i / H_, c = i % H_;
    hfin[(size_t)(b0 + r) * H_ + c] = hF[i];
  }
}

// ---------------------------------------------------------------- kernel 4
__global__ __launch_bounds__(256)
void bn_fc(const float* __restrict__ hfin, const float* __restrict__ gam,
           const float* __restrict__ bet,  const float* __restrict__ bmean,
           const float* __restrict__ bvar, const float* __restrict__ Wfc,
           const float* __restrict__ bfc,  float* __restrict__ out)
{
  const int id = blockIdx.x * blockDim.x + threadIdx.x;   // 0 .. B*O-1
  if (id >= B_ * O_) return;
  const int b = id / O_, o = id % O_;
  float acc = bfc[o];
  for (int c = 0; c < H_; ++c) {
    const float hv  = hfin[(size_t)b * H_ + c];
    const float hbn = (hv - bmean[c]) * rsqrtf(bvar[c] + 1e-5f) * gam[c] + bet[c];
    acc += hbn * Wfc[(size_t)o * H_ + c];
  }
  out[(size_t)b * O_ + o] = acc;
}

// ---------------------------------------------------------------- launch
extern "C" void kernel_launch(void* const* d_in, const int* in_sizes, int n_in,
                              void* d_out, int out_size, void* d_ws, size_t ws_size,
                              hipStream_t stream)
{
  const float* x     = (const float*)d_in[0];
  const float* mask  = (const float*)d_in[1];
  const float* delta = (const float*)d_in[2];
  const float* xmean = (const float*)d_in[3];
  const float* Wih   = (const float*)d_in[4];
  const float* Whh   = (const float*)d_in[5];
  const float* bih   = (const float*)d_in[6];
  const float* bhh   = (const float*)d_in[7];
  const float* gam   = (const float*)d_in[8];
  const float* bet   = (const float*)d_in[9];
  const float* bmean = (const float*)d_in[10];
  const float* bvar  = (const float*)d_in[11];
  const float* Wfc   = (const float*)d_in[12];
  const float* bfc   = (const float*)d_in[13];
  float* out = (float*)d_out;

  char* ws = (char*)d_ws;
  _Float16* cmb   = (_Float16*)ws;                                   // 50,331,648 B
  _Float16* wih16 = (_Float16*)(ws + (size_t)B_ * S_ * K1 * 2);      //    589,824 B
  _Float16* whh16 = (_Float16*)((char*)wih16 + (size_t)N3 * K1 * 2); //  1,572,864 B
  float*    hfin  = (float*)((char*)whh16 + (size_t)N3 * H_ * 2);    //    262,144 B

  impute_pack<<<(B_ * F_ + 255) / 256, 256, 0, stream>>>(x, mask, delta, xmean, cmb);
  cvt_weights<<<(N3 * K1 + N3 * H_ + 255) / 256, 256, 0, stream>>>(Wih, Whh, wih16, whh16);
  grud_recurrence<<<B_ / 16, 256, 0, stream>>>(cmb, wih16, whh16, bih, bhh, hfin);
  bn_fc<<<(B_ * O_ + 255) / 256, 256, 0, stream>>>(hfin, gam, bet, bmean, bvar, Wfc, bfc, out);
}
